// MultiHeadAttention_45749991637390
// MI455X (gfx1250) — matrix operand order repaired and checked by reading, not compiled, not run on hardware
//
#include <hip/hip_runtime.h>

#define H    1024
#define NHD  16
#define HDIM 64
#define SEQ  2048
#define BTCH 2
#define MTOT (BTCH * SEQ)   // 4096

typedef __attribute__((ext_vector_type(16))) __bf16 v16bf;
typedef __attribute__((ext_vector_type(8)))  float  v8f;

union V16 {
    v16bf          bf;
    unsigned short u16[16];
    unsigned int   u32[8];
    uint4          q[2];
};

__device__ __forceinline__ unsigned short f2bf(float f) {
    unsigned int x = __float_as_uint(f);
    x += 0x7fffu + ((x >> 16) & 1u);     // round-to-nearest-even
    return (unsigned short)(x >> 16);
}

// A operand K order chosen here (contiguous, matches B; not AMD's table order): VGPR p holds K pair
//   K = half*16 + 2p
__device__ __forceinline__ int a_pair_k(int p, int half) {
    return half * 16 + 2 * p;
}

__device__ __forceinline__ v16bf load_A(const unsigned short* rowPtr, int half) {
    V16 a;
#pragma unroll
    for (int p = 0; p < 8; ++p) {
        int k = a_pair_k(p, half);
        a.u32[p] = *reinterpret_cast<const unsigned int*>(rowPtr + k);
    }
    return a.bf;
}

// B operand (32x16, 16-bit), this code's contiguous K order: lane = column, elements j -> K = half*16 + j
__device__ __forceinline__ v16bf load_B16(const unsigned short* p) {
    V16 b;
    const uint4* q = reinterpret_cast<const uint4*>(p);
    b.q[0] = q[0];
    b.q[1] = q[1];
    return b.bf;
}

__device__ __forceinline__ v8f wmma_bf16(v16bf a, v16bf b, v8f c) {
    return __builtin_amdgcn_wmma_f32_16x16x32_bf16(false, a, false, b,
                                                   (short)0, c, false, false);
}

// load all tiles of one k-stage into the given buffer slot
__device__ __forceinline__ void load_stage(const unsigned short* arow,
                                           const unsigned short* const brow[4],
                                           int k0, int half,
                                           v16bf& a, v16bf b[4]) {
    a = load_A(arow + k0, half);
#pragma unroll
    for (int t = 0; t < 4; ++t) b[t] = load_B16(brow[t] + k0);
}

// fully unrolled K loop, distance-2 software pipeline with 3-buffer rotation:
// buffer i is loaded at stage s and not reloaded until s+3, so at every WMMA
// the loads of stages s+1 and s+2 remain in flight (partial s_wait_loadcnt).
__device__ __forceinline__ void gemm_k_loop(const unsigned short* arow,
                                            const unsigned short* const brow[4],
                                            int half, v8f acc[4]) {
    const int NS = H / 32;               // 32 stages
    v16bf aB[3];
    v16bf bB[3][4];
    load_stage(arow, brow, 0, half, aB[0], bB[0]);
    load_stage(arow, brow, 32, half, aB[1], bB[1]);

#pragma unroll
    for (int s = 0; s < NS; ++s) {
        if (s + 2 < NS)
            load_stage(arow, brow, (s + 2) * 32, half,
                       aB[(s + 2) % 3], bB[(s + 2) % 3]);
        if ((s & 3) == 0)
            __builtin_prefetch(arow + s * 32 + 256, 0, 1);
#pragma unroll
        for (int t = 0; t < 4; ++t)
            acc[t] = wmma_bf16(aB[s % 3], bB[s % 3][t], acc[t]);
    }
}

// ---------------------------------------------------------------- convert
__global__ void cvt_f32_bf16(const float* __restrict__ in,
                             unsigned short* __restrict__ out, int n) {
    int i = blockIdx.x * blockDim.x + threadIdx.x;
    if (i < n) out[i] = f2bf(in[i]);
}

// ---------------------------------------------------------------- fused QKV
// waves: 256 m-tiles x 48 n-tiles (3H/64). Wave = 16x64 output tile.
__global__ void __launch_bounds__(256)
qkv_gemm(const unsigned short* __restrict__ Xb,
         const unsigned short* __restrict__ Wqb,
         const unsigned short* __restrict__ Wkb,
         const unsigned short* __restrict__ Wvb,
         const float* __restrict__ bq, const float* __restrict__ bk,
         const float* __restrict__ bv,
         unsigned short* __restrict__ Qb, unsigned short* __restrict__ Kb,
         unsigned short* __restrict__ Vtb) {
    const int lane = threadIdx.x & 31;
    const int wid  = (blockIdx.x * blockDim.x + threadIdx.x) >> 5;
    const int NT   = (3 * H) / 64;          // 48
    const int mt = wid / NT, nt = wid % NT;
    const int m0 = mt * 16, n0 = nt * 64;
    const int r = lane & 15, half = lane >> 4;
    const int sel   = n0 / H;               // 0:q 1:k 2:v (wave-uniform)
    const int nbase = n0 % H;
    const unsigned short* W    = (sel == 0) ? Wqb : (sel == 1) ? Wkb : Wvb;
    const float*          bias = (sel == 0) ? bq  : (sel == 1) ? bk  : bv;

    const unsigned short* arow = Xb + (size_t)(m0 + r) * H;
    const unsigned short* brow[4];
#pragma unroll
    for (int t = 0; t < 4; ++t)
        brow[t] = W + (size_t)(nbase + t * 16 + r) * H + half * 16;

    v8f acc[4] = {{}, {}, {}, {}};
    gemm_k_loop(arow, brow, half, acc);

#pragma unroll
    for (int t = 0; t < 4; ++t) {
        int   n  = nbase + t * 16 + r;      // 0..1023 within selected matrix
        float bs = bias[n];
        int   h  = n / HDIM, d = n % HDIM;
#pragma unroll
        for (int p = 0; p < 8; ++p) {
            int   m  = m0 + half * 8 + p;
            int   b_ = m / SEQ, s = m % SEQ;
            float v  = acc[t][p] + bs;
            if (sel == 0) {        // Q pre-scaled by 1/sqrt(HD), [B,NH,S,HD]
                Qb[((size_t)(b_ * NHD + h) * SEQ + s) * HDIM + d] = f2bf(v * 0.125f);
            } else if (sel == 1) { // K [B,NH,S,HD]
                Kb[((size_t)(b_ * NHD + h) * SEQ + s) * HDIM + d] = f2bf(v);
            } else {               // V transposed [B,NH,HD,S]
                Vtb[((size_t)(b_ * NHD + h) * HDIM + d) * SEQ + s] = f2bf(v);
            }
        }
    }
}

// ---------------------------------------------------------------- flash attention
// One 32-key chunk: V tiles (current) + K tiles (next chunk) issued first,
// score WMMAs wait only on kb_cur, softmax VALU overlaps in-flight loads.
__device__ __forceinline__ void flash_step(int kc, int kn,
                                           const unsigned short* kbase,
                                           const unsigned short* vbase,
                                           int r, int half,
                                           unsigned short (&pTile)[16][32],
                                           const v16bf& qa0, const v16bf& qa1,
                                           const v16bf kb_cur[2][2],
                                           v16bf kb_nxt[2][2],
                                           float m_run[8], float l_run[8],
                                           v8f acc[4]) {
    // V tiles for current chunk (consumed only after softmax)
    v16bf vb[4];
#pragma unroll
    for (int c = 0; c < 4; ++c)
        vb[c] = load_B16(vbase + (size_t)(c * 16 + r) * SEQ + kc + half * 16);
    // K tiles for next chunk
#pragma unroll
    for (int t = 0; t < 2; ++t)
#pragma unroll
        for (int c = 0; c < 2; ++c)
            kb_nxt[t][c] = load_B16(kbase + (size_t)(kn + t * 16 + r) * HDIM +
                                    c * 32 + half * 16);

    // S = Q K^T  (Q pre-scaled by 1/sqrt(HD))
    v8f s0 = {}, s1 = {};
    s0 = wmma_bf16(qa0, kb_cur[0][0], s0);
    s0 = wmma_bf16(qa1, kb_cur[0][1], s0);
    s1 = wmma_bf16(qa0, kb_cur[1][0], s1);
    s1 = wmma_bf16(qa1, kb_cur[1][1], s1);

    // online softmax: rows of the D tile live inside one 16-lane half
    float pe0[8], pe1[8], alpha[8];
#pragma unroll
    for (int p = 0; p < 8; ++p) {
        float mx = fmaxf(s0[p], s1[p]);
        mx = fmaxf(mx, __shfl_xor(mx, 1, 32));
        mx = fmaxf(mx, __shfl_xor(mx, 2, 32));
        mx = fmaxf(mx, __shfl_xor(mx, 4, 32));
        mx = fmaxf(mx, __shfl_xor(mx, 8, 32));
        float mnew = fmaxf(m_run[p], mx);
        float al   = __expf(m_run[p] - mnew);
        m_run[p] = mnew;
        alpha[p] = al;
        float e0 = __expf(s0[p] - mnew);
        float e1 = __expf(s1[p] - mnew);
        pe0[p] = e0;
        pe1[p] = e1;
        float rs = e0 + e1;
        rs += __shfl_xor(rs, 1, 32);
        rs += __shfl_xor(rs, 2, 32);
        rs += __shfl_xor(rs, 4, 32);
        rs += __shfl_xor(rs, 8, 32);
        l_run[p] = l_run[p] * al + rs;
    }
#pragma unroll
    for (int c = 0; c < 4; ++c)
#pragma unroll
        for (int p = 0; p < 8; ++p) acc[c][p] *= alpha[p];

    // stage P (D-layout) through LDS, reload in A-layout (wave-private tile,
    // DS ops are in-order within a wave -> no barrier needed)
#pragma unroll
    for (int p = 0; p < 8; ++p) {
        int row = half * 8 + p;
        pTile[row][r]      = f2bf(pe0[p]);
        pTile[row][16 + r] = f2bf(pe1[p]);
    }
    V16 pa;
    const unsigned short* prow = &pTile[r][0];
#pragma unroll
    for (int p = 0; p < 8; ++p) {
        int k = a_pair_k(p, half);
        pa.u32[p] = *reinterpret_cast<const unsigned int*>(prow + k);
    }

    // ctx += P @ V   (waits only on vb; kb_nxt still in flight)
#pragma unroll
    for (int c = 0; c < 4; ++c) acc[c] = wmma_bf16(pa.bf, vb[c], acc[c]);
}

__global__ void __launch_bounds__(256)
flash_attn(const unsigned short* __restrict__ Qb,
           const unsigned short* __restrict__ Kb,
           const unsigned short* __restrict__ Vtb,
           unsigned short* __restrict__ Ctxb) {
    __shared__ unsigned short ldsP[8][16][32];   // per-wave P staging (D->A relayout)
    const int lane  = threadIdx.x & 31;
    const int wslot = threadIdx.x >> 5;
    const int wid   = (blockIdx.x * blockDim.x + threadIdx.x) >> 5;
    const int QT    = SEQ / 16;                  // 128
    const int bh = wid / QT;
    const int q0 = (wid % QT) * 16;
    const int r = lane & 15, half = lane >> 4;
    const int b_ = bh / NHD, h = bh % NHD;

    const unsigned short* qrow = Qb + ((size_t)bh * SEQ + q0 + r) * HDIM;
    v16bf qa0 = load_A(qrow + 0, half);          // d = 0..31
    v16bf qa1 = load_A(qrow + 32, half);         // d = 32..63

    const unsigned short* kbase = Kb + (size_t)bh * SEQ * HDIM;
    const unsigned short* vbase = Vtb + (size_t)bh * HDIM * SEQ;

    float m_run[8], l_run[8];
    v8f   acc[4] = {{}, {}, {}, {}};
#pragma unroll
    for (int p = 0; p < 8; ++p) { m_run[p] = -3.0e38f; l_run[p] = 0.f; }

    // prologue: K tiles for chunk 0 into buffer A
    v16bf kbA[2][2], kbB[2][2];
#pragma unroll
    for (int t = 0; t < 2; ++t)
#pragma unroll
        for (int c = 0; c < 2; ++c)
            kbA[t][c] =
                load_B16(kbase + (size_t)(t * 16 + r) * HDIM + c * 32 + half * 16);

    // ping-pong over chunks, unrolled x2 (SEQ/32 = 64 chunks, even)
    for (int kc = 0; kc < SEQ; kc += 64) {
        flash_step(kc, kc + 32, kbase, vbase, r, half, ldsP[wslot],
                   qa0, qa1, kbA, kbB, m_run, l_run, acc);
        flash_step(kc + 32, (kc + 64) & (SEQ - 1), kbase, vbase, r, half,
                   ldsP[wslot], qa0, qa1, kbB, kbA, m_run, l_run, acc);
    }

    // normalize, store ctx bf16 in [B, S, NH*HD] merged layout
#pragma unroll
    for (int p = 0; p < 8; ++p) {
        float inv = 1.0f / l_run[p];
        int   s   = q0 + half * 8 + p;
        unsigned short* crow = Ctxb + ((size_t)b_ * SEQ + s) * H + h * HDIM;
#pragma unroll
        for (int c = 0; c < 4; ++c) crow[c * 16 + r] = f2bf(acc[c][p] * inv);
    }
}

// ---------------------------------------------------------------- output proj
__global__ void __launch_bounds__(256)
out_gemm(const unsigned short* __restrict__ Ctxb,
         const unsigned short* __restrict__ Wob,
         const float* __restrict__ bo, float* __restrict__ Out) {
    const int lane = threadIdx.x & 31;
    const int wid  = (blockIdx.x * blockDim.x + threadIdx.x) >> 5;
    const int NT   = H / 64;                // 16
    const int mt = wid / NT, nt = wid % NT;
    const int m0 = mt * 16, n0 = nt * 64;
    const int r = lane & 15, half = lane >> 4;

    const unsigned short* arow = Ctxb + (size_t)(m0 + r) * H;
    const unsigned short* brow[4];
#pragma unroll
    for (int t = 0; t < 4; ++t)
        brow[t] = Wob + (size_t)(n0 + t * 16 + r) * H + half * 16;

    v8f acc[4] = {{}, {}, {}, {}};
    gemm_k_loop(arow, brow, half, acc);

#pragma unroll
    for (int t = 0; t < 4; ++t) {
        int   n    = n0 + t * 16 + r;
        float bias = bo[n];
#pragma unroll
        for (int p = 0; p < 8; ++p) {
            int m = m0 + half * 8 + p;
            Out[(size_t)m * H + n] = acc[t][p] + bias;
        }
    }
}

// ---------------------------------------------------------------- launch
extern "C" void kernel_launch(void* const* d_in, const int* in_sizes, int n_in,
                              void* d_out, int out_size, void* d_ws, size_t ws_size,
                              hipStream_t stream) {
    (void)in_sizes; (void)n_in; (void)out_size; (void)ws_size;
    const float* hs = (const float*)d_in[0];
    const float* Wq = (const float*)d_in[1];
    const float* bq = (const float*)d_in[2];
    const float* Wk = (const float*)d_in[3];
    const float* bk = (const float*)d_in[4];
    const float* Wv = (const float*)d_in[5];
    const float* bv = (const float*)d_in[6];
    const float* Wo = (const float*)d_in[7];
    const float* bo = (const float*)d_in[8];
    float* out = (float*)d_out;

    char*  ws  = (char*)d_ws;
    size_t off = 0;
    unsigned short* Xb   = (unsigned short*)(ws + off); off += (size_t)MTOT * H * 2;
    unsigned short* Wqb  = (unsigned short*)(ws + off); off += (size_t)H * H * 2;
    unsigned short* Wkb  = (unsigned short*)(ws + off); off += (size_t)H * H * 2;
    unsigned short* Wvb  = (unsigned short*)(ws + off); off += (size_t)H * H * 2;
    unsigned short* Wob  = (unsigned short*)(ws + off); off += (size_t)H * H * 2;
    unsigned short* Qb   = (unsigned short*)(ws + off); off += (size_t)MTOT * H * 2;
    unsigned short* Kb   = (unsigned short*)(ws + off); off += (size_t)MTOT * H * 2;
    unsigned short* Vtb  = (unsigned short*)(ws + off); off += (size_t)MTOT * H * 2;
    unsigned short* Ctxb = (unsigned short*)(ws + off); off += (size_t)MTOT * H * 2;
    // total workspace: 48 MB

    const int nAct = MTOT * H;   // 4M elements
    const int nW   = H * H;      // 1M elements
    cvt_f32_bf16<<<(nAct + 255) / 256, 256, 0, stream>>>(hs, Xb, nAct);
    cvt_f32_bf16<<<(nW + 255) / 256, 256, 0, stream>>>(Wq, Wqb, nW);
    cvt_f32_bf16<<<(nW + 255) / 256, 256, 0, stream>>>(Wk, Wkb, nW);
    cvt_f32_bf16<<<(nW + 255) / 256, 256, 0, stream>>>(Wv, Wvb, nW);
    cvt_f32_bf16<<<(nW + 255) / 256, 256, 0, stream>>>(Wo, Wob, nW);

    // 256 m-tiles * 48 n-tiles = 12288 waves / 8 waves-per-block
    qkv_gemm<<<1536, 256, 0, stream>>>(Xb, Wqb, Wkb, Wvb, bq, bk, bv, Qb, Kb, Vtb);
    // 32 (b,h) * 128 q-tiles = 4096 waves / 8
    flash_attn<<<512, 256, 0, stream>>>(Qb, Kb, Vtb, Ctxb);
    // 256 m-tiles * 16 n-tiles = 4096 waves / 8
    out_gemm<<<512, 256, 0, stream>>>(Ctxb, Wob, bo, out);
}
